// LocalGlobalAttention_19464791785685
// MI455X (gfx1250) — compile-verified, matrix operand order checked by reading, not run on hardware
//
#include <hip/hip_runtime.h>
#include <hip/hip_fp16.h>

typedef __attribute__((ext_vector_type(16))) _Float16 v16h;
typedef __attribute__((ext_vector_type(8)))  _Float16 v8h;
typedef __attribute__((ext_vector_type(8)))  float    v8f;

#define DIM     768
#define HEADS   12
#define HD      64
#define SEQ     2048
#define BATCH   2
#define MROWS   (BATCH*SEQ)       // 4096
#define KSTEPS  (DIM/32)          // 24 wmma K-steps per GEMM dot
#define NT64    (DIM/64)          // 12 64-wide N tiles (== heads)
#define MT      (MROWS/16)        // 256 M tiles

// Load a 16x32-f16 A/B fragment for this lane.  `base` already includes the
// lane's row/col offset plus kb = 8*(lane>>4).  Element e of the v16h maps to
// K = e + 8*(e>>3) + 8*(lane>>4), i.e. two contiguous 16-byte chunks.
__device__ __forceinline__ v16h frag16(const _Float16* base) {
  v8h lo = *(const v8h*)(base);
  v8h hi = *(const v8h*)(base + 16);
  v16h r;
#pragma unroll
  for (int i = 0; i < 8; ++i) { r[i] = lo[i]; r[8 + i] = hi[i]; }
  return r;
}

// ---------------- prep kernels ----------------
__global__ void lga_cvt_f16(const float* __restrict__ in, _Float16* __restrict__ out, int n4) {
  int i = blockIdx.x * blockDim.x + threadIdx.x;
  if (i < n4) {
    float4 v = *(const float4*)(in + i * 4);
    _Float16* o = out + i * 4;
    o[0] = (_Float16)v.x; o[1] = (_Float16)v.y; o[2] = (_Float16)v.z; o[3] = (_Float16)v.w;
  }
}

__global__ void lga_transpose_w(const float* __restrict__ W, _Float16* __restrict__ WT) {
  int idx = blockIdx.x * blockDim.x + threadIdx.x;          // n-major index
  if (idx < DIM * DIM) {
    int n = idx / DIM, k = idx - n * DIM;
    WT[idx] = (_Float16)W[k * DIM + n];                     // WT[n][k] = W[k][n]
  }
}

// 16x64 per-wave GEMM core: one A fragment feeds 4 WMMAs per K-step.
// Software-pipelined: fragments for step kt+1 are loaded (into separate
// loop-carried registers) while step kt's 4 WMMAs execute, so the loads are
// issued one full XDL group ahead of their first use.
__device__ __forceinline__ void gemm16x64(
    const _Float16* __restrict__ arow,     // + row*DIM + kb
    const _Float16* __restrict__ b0, const _Float16* __restrict__ b1,
    const _Float16* __restrict__ b2, const _Float16* __restrict__ b3,
    v8f& c0, v8f& c1, v8f& c2, v8f& c3)
{
  v16h a  = frag16(arow);
  v16h f0 = frag16(b0);
  v16h f1 = frag16(b1);
  v16h f2 = frag16(b2);
  v16h f3 = frag16(b3);
#pragma unroll 4
  for (int kt = 0; kt < KSTEPS - 1; ++kt) {
    const int o = (kt + 1) * 32;
    __builtin_prefetch(arow + o + 64, 0, 3);   // 2 steps ahead, near scope
    __builtin_prefetch(b0   + o + 64, 0, 3);
    __builtin_prefetch(b1   + o + 64, 0, 3);
    __builtin_prefetch(b2   + o + 64, 0, 3);
    __builtin_prefetch(b3   + o + 64, 0, 3);
    v16h an = frag16(arow + o);                // next-step fragments
    v16h g0 = frag16(b0 + o);
    v16h g1 = frag16(b1 + o);
    v16h g2 = frag16(b2 + o);
    v16h g3 = frag16(b3 + o);
    c0 = __builtin_amdgcn_wmma_f32_16x16x32_f16(false, a, false, f0, (short)0, c0, false, false);
    c1 = __builtin_amdgcn_wmma_f32_16x16x32_f16(false, a, false, f1, (short)0, c1, false, false);
    c2 = __builtin_amdgcn_wmma_f32_16x16x32_f16(false, a, false, f2, (short)0, c2, false, false);
    c3 = __builtin_amdgcn_wmma_f32_16x16x32_f16(false, a, false, f3, (short)0, c3, false, false);
    a = an; f0 = g0; f1 = g1; f2 = g2; f3 = g3;
  }
  // peeled final step
  c0 = __builtin_amdgcn_wmma_f32_16x16x32_f16(false, a, false, f0, (short)0, c0, false, false);
  c1 = __builtin_amdgcn_wmma_f32_16x16x32_f16(false, a, false, f1, (short)0, c1, false, false);
  c2 = __builtin_amdgcn_wmma_f32_16x16x32_f16(false, a, false, f2, (short)0, c2, false, false);
  c3 = __builtin_amdgcn_wmma_f32_16x16x32_f16(false, a, false, f3, (short)0, c3, false, false);
}

// ---------------- fused QKV projection ----------------
// grid.x = MT*NT64/8 tiles (8 waves/block, 16x64 tile per wave), grid.z = 0:Q 1:K 2:V
__global__ void __launch_bounds__(256) lga_qkv_gemm(
    const _Float16* __restrict__ xh,
    const _Float16* __restrict__ WqT, const _Float16* __restrict__ WkT, const _Float16* __restrict__ WvT,
    const float* __restrict__ bq, const float* __restrict__ bk, const float* __restrict__ bv,
    _Float16* __restrict__ Qb, _Float16* __restrict__ Kb, _Float16* __restrict__ Vt)
{
  const int wid = threadIdx.x >> 5, lane = threadIdx.x & 31;
  const int tile = blockIdx.x * 8 + wid;
  const int mt = tile / NT64, n64 = tile - mt * NT64;      // n64 == head index
  const int which = blockIdx.z;
  const _Float16* WT = (which == 0) ? WqT : (which == 1) ? WkT : WvT;
  const float* bias  = (which == 0) ? bq  : (which == 1) ? bk  : bv;
  const int lm = lane & 15, lh = lane >> 4, kb = lh * 8;

  const _Float16* arow = xh + (size_t)(mt * 16 + lm) * DIM + kb;
  const _Float16* bb   = WT + (size_t)(n64 * 64 + lm) * DIM + kb;
  v8f c0 = {}, c1 = {}, c2 = {}, c3 = {};
  gemm16x64(arow, bb, bb + 16 * DIM, bb + 32 * DIM, bb + 48 * DIM, c0, c1, c2, c3);

  const v8f* cs[4] = { &c0, &c1, &c2, &c3 };
#pragma unroll
  for (int j = 0; j < 4; ++j) {
    const int d = j * 16 + lm;                // head-dim index (n64 is the head)
    const float bn = bias[n64 * 64 + d];
#pragma unroll
    for (int r = 0; r < 8; ++r) {
      int m = mt * 16 + r + lh * 8;
      int bi = m >> 11, s = m & (SEQ - 1);
      float val = (*cs[j])[r] + bn;
      size_t hb = (size_t)(bi * HEADS + n64);
      if (which == 2)      Vt[(hb * HD + d) * SEQ + s] = (_Float16)val;  // V transposed [b,h,d,s]
      else if (which == 0) Qb[(hb * SEQ + s) * HD + d] = (_Float16)val;  // [b,h,s,d]
      else                 Kb[(hb * SEQ + s) * HD + d] = (_Float16)val;
    }
  }
}

// ---------------- local+global attention ----------------
// One wave per (b,h, 16-query tile). 4 waves / block (LDS: 4 * 15360 B).
__global__ void __launch_bounds__(128) lga_attn(
    const _Float16* __restrict__ Qb, const _Float16* __restrict__ Kb,
    const _Float16* __restrict__ Vt, _Float16* __restrict__ AO)
{
  __shared__ __align__(16) _Float16 smem[4 * 7680];   // per wave: P 16x96 + V' 64x96
  const int wid = threadIdx.x >> 5, lane = threadIdx.x & 31;
  const int tile = blockIdx.x * 4 + wid;
  const int bh = tile >> 7;                 // / (SEQ/16)
  const int qt = tile & 127;
  const int q0 = qt * 16;
  const int lm = lane & 15, lh = lane >> 4, kb = lh * 8;

  const _Float16* Qh = Qb + (size_t)bh * SEQ * HD;
  const _Float16* Kh = Kb + (size_t)bh * SEQ * HD;
  const _Float16* Vh = Vt + (size_t)bh * HD * SEQ;
  _Float16* P  = smem + wid * 7680;         // [16][96]
  _Float16* Vb = P + 16 * 96;               // [64][96]

  // Q fragments, reused across all key tiles (head_dim 64 = 2 K-steps)
  const _Float16* qrow = Qh + (size_t)(q0 + lm) * HD + kb;
  const v16h a0 = frag16(qrow);
  const v16h a1 = frag16(qrow + 32);

  const int kstart = q0 - 32;               // band covers kstart..kstart+79
  float sc[6][8];
#pragma unroll
  for (int t = 0; t < 6; ++t) {             // t=0..4 local band, t=5 global keys 0..3
    int ks = (t < 5) ? (kstart + t * 16) : 0;
    int srow = ks + lm;
    int src = min(max(srow, 0), SEQ - 1);   // clamp; masked later
    const _Float16* kcol = Kh + (size_t)src * HD + kb;
    v16h b0 = frag16(kcol);
    v16h b1 = frag16(kcol + 32);
    v8f c = {};
    c = __builtin_amdgcn_wmma_f32_16x16x32_f16(false, a0, false, b0, (short)0, c, false, false);
    c = __builtin_amdgcn_wmma_f32_16x16x32_f16(false, a1, false, b1, (short)0, c, false, false);
#pragma unroll
    for (int r = 0; r < 8; ++r) sc[t][r] = c[r];
  }

  const float scale = 0.125f;               // 64^-0.5
#pragma unroll
  for (int r = 0; r < 8; ++r) {
    int qs = q0 + r + lh * 8;
    // ---- local softmax over the 80-key band ----
    float v[5], mx = -1e30f;
#pragma unroll
    for (int t = 0; t < 5; ++t) {
      int ksr = kstart + t * 16 + lm;
      int dd = qs - ksr;
      bool ok = (ksr >= 0) && (ksr < SEQ) && (dd <= 32) && (dd >= -32);
      v[t] = ok ? sc[t][r] * scale : -1e30f;
      mx = fmaxf(mx, v[t]);
    }
    for (int m_ = 1; m_ < 16; m_ <<= 1) mx = fmaxf(mx, __shfl_xor(mx, m_, 32));
    float sum = 0.f;
#pragma unroll
    for (int t = 0; t < 5; ++t) { v[t] = __expf(v[t] - mx); sum += v[t]; }
    for (int m_ = 1; m_ < 16; m_ <<= 1) sum += __shfl_xor(sum, m_, 32);
    float inv = 0.5f / sum;                 // fold the 0.5*local weight in
#pragma unroll
    for (int t = 0; t < 5; ++t) sc[t][r] = v[t] * inv;
    // ---- global softmax over keys 0..3 (top_k of softmax(g) with k==N is all;
    //      attention over a key set is permutation invariant, so g is moot) ----
    float gv = (lm < 4) ? sc[5][r] * scale : -1e30f;
    float gm = gv;
    for (int m_ = 1; m_ < 16; m_ <<= 1) gm = fmaxf(gm, __shfl_xor(gm, m_, 32));
    float ge = __expf(gv - gm);
    float gs = ge;
    for (int m_ = 1; m_ < 16; m_ <<= 1) gs += __shfl_xor(gs, m_, 32);
    sc[5][r] = ge * (0.5f / gs);
  }

  // P -> LDS in A-fragment source layout [q][k0..95] (cols 80..95 = global, p=0 where invalid)
#pragma unroll
  for (int t = 0; t < 6; ++t)
#pragma unroll
    for (int r = 0; r < 8; ++r)
      P[(r + lh * 8) * 96 + t * 16 + lm] = (_Float16)sc[t][r];

  // V' -> LDS: rows 0..79 = band (zero out-of-range), rows 80..95 = keys 0..15 (only 0..3 weighted)
  for (int idx = lane; idx < 64 * 96; idx += 32) {
    int d = idx / 96, k = idx - d * 96;
    _Float16 val;
    if (k < 80) {
      int s = kstart + k;
      val = (s >= 0 && s < SEQ) ? Vh[(size_t)d * SEQ + s] : (_Float16)0.f;
    } else {
      val = Vh[(size_t)d * SEQ + (k - 80)];
    }
    Vb[idx] = val;
  }
  __syncthreads();

  // out[16q x 64d] = P(16x96) @ V'(96x64)
  const int b_ = bh / HEADS, h_ = bh - b_ * HEADS;
#pragma unroll
  for (int ntd = 0; ntd < 4; ++ntd) {
    v8f c = {};
#pragma unroll
    for (int ks = 0; ks < 3; ++ks) {
      v16h a = frag16(P + lm * 96 + ks * 32 + kb);
      v16h b = frag16(Vb + (ntd * 16 + lm) * 96 + ks * 32 + kb);
      c = __builtin_amdgcn_wmma_f32_16x16x32_f16(false, a, false, b, (short)0, c, false, false);
    }
    int n = h_ * HD + ntd * 16 + lm;
#pragma unroll
    for (int r = 0; r < 8; ++r) {
      int m = b_ * SEQ + q0 + r + lh * 8;
      AO[(size_t)m * DIM + n] = (_Float16)c[r];
    }
  }
}

// ---------------- output projection (16x64 tile per wave) ----------------
__global__ void __launch_bounds__(256) lga_out_gemm(
    const _Float16* __restrict__ AO, const _Float16* __restrict__ WoT,
    const float* __restrict__ bo, float* __restrict__ out)
{
  const int wid = threadIdx.x >> 5, lane = threadIdx.x & 31;
  const int tile = blockIdx.x * 8 + wid;
  const int mt = tile / NT64, n64 = tile - mt * NT64;
  const int lm = lane & 15, lh = lane >> 4, kb = lh * 8;

  const _Float16* arow = AO  + (size_t)(mt * 16 + lm) * DIM + kb;
  const _Float16* bb   = WoT + (size_t)(n64 * 64 + lm) * DIM + kb;
  v8f c0 = {}, c1 = {}, c2 = {}, c3 = {};
  gemm16x64(arow, bb, bb + 16 * DIM, bb + 32 * DIM, bb + 48 * DIM, c0, c1, c2, c3);

  const v8f* cs[4] = { &c0, &c1, &c2, &c3 };
#pragma unroll
  for (int j = 0; j < 4; ++j) {
    const int n = n64 * 64 + j * 16 + lm;
    const float bn = bo[n];
#pragma unroll
    for (int r = 0; r < 8; ++r) {
      int m = mt * 16 + r + lh * 8;
      out[(size_t)m * DIM + n] = (*cs[j])[r] + bn;
    }
  }
}

extern "C" void kernel_launch(void* const* d_in, const int* in_sizes, int n_in,
                              void* d_out, int out_size, void* d_ws, size_t ws_size,
                              hipStream_t stream) {
  (void)in_sizes; (void)n_in; (void)out_size; (void)ws_size;
  const float* x  = (const float*)d_in[0];
  const float* Wq = (const float*)d_in[1];
  const float* bq = (const float*)d_in[2];
  const float* Wk = (const float*)d_in[3];
  const float* bk = (const float*)d_in[4];
  const float* Wv = (const float*)d_in[5];
  const float* bv = (const float*)d_in[6];
  const float* Wo = (const float*)d_in[7];
  const float* bo = (const float*)d_in[8];
  // d_in[9] = g: unused (see comment in lga_attn)
  float* out = (float*)d_out;

  _Float16* ws  = (_Float16*)d_ws;
  _Float16* xh  = ws;                         // 4096*768
  _Float16* WqT = xh  + (size_t)MROWS * DIM;  // 768*768 each
  _Float16* WkT = WqT + (size_t)DIM * DIM;
  _Float16* WvT = WkT + (size_t)DIM * DIM;
  _Float16* WoT = WvT + (size_t)DIM * DIM;
  _Float16* Qb  = WoT + (size_t)DIM * DIM;    // 4096*768 each
  _Float16* Kb  = Qb  + (size_t)MROWS * DIM;
  _Float16* Vt  = Kb  + (size_t)MROWS * DIM;
  _Float16* AO  = Vt  + (size_t)MROWS * DIM;

  // prep: convert x, transpose+convert weights
  lga_cvt_f16<<<(MROWS * DIM / 4 + 255) / 256, 256, 0, stream>>>(x, xh, MROWS * DIM / 4);
  const int tw = (DIM * DIM + 255) / 256;
  lga_transpose_w<<<tw, 256, 0, stream>>>(Wq, WqT);
  lga_transpose_w<<<tw, 256, 0, stream>>>(Wk, WkT);
  lga_transpose_w<<<tw, 256, 0, stream>>>(Wv, WvT);
  lga_transpose_w<<<tw, 256, 0, stream>>>(Wo, WoT);

  // QKV projection (z selects Q/K/V): MT*NT64 = 3072 wave-tiles, 8 waves/block
  dim3 gq(MT * NT64 / 8, 1, 3);
  lga_qkv_gemm<<<gq, 256, 0, stream>>>(xh, WqT, WkT, WvT, bq, bk, bv, Qb, Kb, Vt);

  // attention: BATCH*HEADS*(SEQ/16) = 3072 wave-tiles, 4 waves/block
  lga_attn<<<BATCH * HEADS * (SEQ / 16) / 4, 128, 0, stream>>>(Qb, Kb, Vt, AO);

  // output projection
  lga_out_gemm<<<MT * NT64 / 8, 256, 0, stream>>>(AO, WoT, bo, out);
}
